// ROIPooling_68891275427984
// MI455X (gfx1250) — matrix operand order repaired and checked by reading, not compiled, not run on hardware
//
#include <hip/hip_runtime.h>
#include <hip/hip_bf16.h>

// ---------- types ----------
typedef __attribute__((ext_vector_type(16))) __bf16 v16bf;
typedef __attribute__((ext_vector_type(8)))  __bf16 v8bf;
typedef __attribute__((ext_vector_type(4)))  __bf16 v4bf;
typedef __attribute__((ext_vector_type(8)))  float  v8f;

// ---------- problem constants ----------
#define C_IN   1024
#define C1_    256
#define C2_    128
#define OUTD   2048
#define BATCH  8
#define NBOX   64
#define NROI   (BATCH * NBOX)   // 512
#define FH     64
#define FW     64
#define M1     (NROI * 16)      // 8192 rows for GEMM1/GEMM2
#define K3     (C2_ * 16)       // 2048

__device__ __forceinline__ float eluf(float v) {
  return v > 0.0f ? v : __expf(v) - 1.0f;
}

// ---------- WMMA fragment loaders (contiguous-K order k = 16*half + elem, wave32) ----------
// A tile 16(M) x 32(K), bf16, from row-major [M][K]:
//   lanes 0-15  : row = lane,    elems 0..15 = K 0..15
//   lanes 16-31 : row = lane-16, elems 0..15 = K 16..31
__device__ __forceinline__ v16bf load_a_frag(const __bf16* Arow, int ld, int k0, int lane) {
  const int hf = lane >> 4;
  const int r  = lane & 15;
  const __bf16* p = Arow + r * ld + k0 + 16 * hf;
  v8bf lo = *(const v8bf*)(p);
  v8bf hi = *(const v8bf*)(p + 8);
  v16bf f;
#pragma unroll
  for (int i = 0; i < 8; ++i) { f[i] = lo[i]; f[i + 8] = hi[i]; }
  return f;
}

// B tile 32(K) x 16(N), bf16, weights stored as Bt[n][k] row-major (contiguous K):
//   lanes 0-15 hold K 0..15 of column n=lane, lanes 16-31 hold K 16..31.
__device__ __forceinline__ v16bf load_b_frag(const __bf16* Bt, int ld, int k0, int lane) {
  const int hf = lane >> 4;
  const int n  = lane & 15;
  return *(const v16bf*)(Bt + n * ld + k0 + 16 * hf);
}

// One 16x64 output tile, software-pipelined (2-stage register ping-pong):
// loads for stage k+32 are issued before the WMMAs consuming stage k, so the
// wave keeps ~10 loads in flight across every WMMA group instead of draining
// loadcnt to 0 each step. K must be a multiple of 64 (1024/256/2048 all are).
__device__ __forceinline__ void wmma_16x64(const __bf16* Arow, const __bf16* Bt0,
                                           int K, int lane, v8f acc[4]) {
  v16bf a0 = load_a_frag(Arow, K, 0, lane);
  v16bf b0[4];
#pragma unroll
  for (int j = 0; j < 4; ++j) b0[j] = load_b_frag(Bt0 + (size_t)j * 16 * K, K, 0, lane);

  for (int k0 = 0; k0 < K; k0 += 64) {
    // prefetch stage k0+32 while stage k0 computes
    v16bf a1 = load_a_frag(Arow, K, k0 + 32, lane);
    v16bf b1[4];
#pragma unroll
    for (int j = 0; j < 4; ++j) b1[j] = load_b_frag(Bt0 + (size_t)j * 16 * K, K, k0 + 32, lane);

#pragma unroll
    for (int j = 0; j < 4; ++j)
      acc[j] = __builtin_amdgcn_wmma_f32_16x16x32_bf16(
          false, a0, false, b0[j], (short)0, acc[j], false, false);

    // prefetch stage k0+64 (skipped on last iteration; uniform branch)
    if (k0 + 64 < K) {
      a0 = load_a_frag(Arow, K, k0 + 64, lane);
#pragma unroll
      for (int j = 0; j < 4; ++j) b0[j] = load_b_frag(Bt0 + (size_t)j * 16 * K, K, k0 + 64, lane);
    }

#pragma unroll
    for (int j = 0; j < 4; ++j)
      acc[j] = __builtin_amdgcn_wmma_f32_16x16x32_bf16(
          false, a1, false, b1[j], (short)0, acc[j], false, false);
  }
}

// ---------- stage 0: f32 -> bf16 weight conversion ----------
__global__ void cvt_bf16_kernel(const float* __restrict__ in, __bf16* __restrict__ out, int n) {
  int i = blockIdx.x * 256 + threadIdx.x;
  if (i < n) out[i] = (__bf16)in[i];
}

// ---------- stage 1: NCHW f32 -> NHWC bf16 (LDS-tiled transpose) ----------
// grid: (FW/32, C_IN/32, BATCH*FH), block: (32, 8)
__global__ void transpose_nhwc_kernel(const float* __restrict__ x, __bf16* __restrict__ xt) {
  __shared__ __bf16 tile[32][33];
  const int bh = blockIdx.z;
  const int b  = bh >> 6;
  const int h  = bh & 63;
  const int w0 = blockIdx.x * 32;
  const int c0 = blockIdx.y * 32;
  const int tx = threadIdx.x;   // 0..31 -> along W on read, along C on write
  const int ty = threadIdx.y;   // 0..7
#pragma unroll
  for (int i = 0; i < 4; ++i) {
    int c = c0 + ty + 8 * i;
    tile[ty + 8 * i][tx] = (__bf16)x[(((size_t)b * C_IN + c) * FH + h) * FW + w0 + tx];
  }
  __syncthreads();
#pragma unroll
  for (int i = 0; i < 4; ++i) {
    int w = w0 + ty + 8 * i;
    xt[(((size_t)b * FH + h) * FW + w) * C_IN + c0 + tx] = tile[tx][ty + 8 * i];
  }
}

// ---------- stage 2: ROI-align -> A1[m = r*16 + ph*4 + pw][c] bf16 ----------
// grid: M1 blocks, 256 threads, 4 channels per thread (coalesced 8B vector taps)
__global__ void roi_align_kernel(const __bf16* __restrict__ xt,
                                 const float* __restrict__ boxes,
                                 const float* __restrict__ scale_p,
                                 __bf16* __restrict__ A1) {
  const int m  = blockIdx.x;
  const int r  = m >> 4;
  const int p  = m & 15;
  const int ph = p >> 2;
  const int pw = p & 3;
  const int b  = r >> 6;          // NBOX = 64
  const int nb = r & 63;

  const float sc = scale_p[0];
  const float4 bx = *(const float4*)(boxes + ((size_t)b * NBOX + nb) * 4);
  const float x1 = bx.x * sc, y1 = bx.y * sc, x2 = bx.z * sc, y2 = bx.w * sc;
  const float rw = fmaxf(x2 - x1, 1.0f);
  const float rh = fmaxf(y2 - y1, 1.0f);
  const float binw = rw * 0.25f;
  const float binh = rh * 0.25f;

  const int c0 = threadIdx.x * 4;
  float acc[4] = {0.f, 0.f, 0.f, 0.f};

#pragma unroll
  for (int sy = 0; sy < 2; ++sy) {
    const int   jy   = 2 * ph + sy;
    const float offy = (float)(jy >> 1) + ((float)(jy & 1) + 0.5f) * 0.5f;
    float cy = y1 + offy * binh;
    const bool vy = (cy >= -1.0f) && (cy <= (float)FH);
    cy = fminf(fmaxf(cy, 0.0f), (float)(FH - 1));
    int ylo = min((int)floorf(cy), FH - 1);
    int yhi = min(ylo + 1, FH - 1);
    const float ly = cy - (float)ylo;
    const float hy = 1.0f - ly;
#pragma unroll
    for (int sx = 0; sx < 2; ++sx) {
      const int   jx   = 2 * pw + sx;
      const float offx = (float)(jx >> 1) + ((float)(jx & 1) + 0.5f) * 0.5f;
      float cx = x1 + offx * binw;
      const bool vx = (cx >= -1.0f) && (cx <= (float)FW);
      cx = fminf(fmaxf(cx, 0.0f), (float)(FW - 1));
      int xlo = min((int)floorf(cx), FW - 1);
      int xhi = min(xlo + 1, FW - 1);
      const float lx = cx - (float)xlo;
      const float hx = 1.0f - lx;
      if (vy && vx) {
        const size_t rowlo = (((size_t)b * FH + ylo) * FW) * C_IN;
        const size_t rowhi = (((size_t)b * FH + yhi) * FW) * C_IN;
        v4bf t00 = *(const v4bf*)(xt + rowlo + (size_t)xlo * C_IN + c0);
        v4bf t01 = *(const v4bf*)(xt + rowlo + (size_t)xhi * C_IN + c0);
        v4bf t10 = *(const v4bf*)(xt + rowhi + (size_t)xlo * C_IN + c0);
        v4bf t11 = *(const v4bf*)(xt + rowhi + (size_t)xhi * C_IN + c0);
        const float w00 = hy * hx, w01 = hy * lx, w10 = ly * hx, w11 = ly * lx;
#pragma unroll
        for (int i = 0; i < 4; ++i) {
          acc[i] += w00 * (float)t00[i] + w01 * (float)t01[i] +
                    w10 * (float)t10[i] + w11 * (float)t11[i];
        }
      }
    }
  }
  v4bf out;
#pragma unroll
  for (int i = 0; i < 4; ++i) out[i] = (__bf16)(acc[i] * 0.25f);
  *(v4bf*)(A1 + (size_t)m * C_IN + c0) = out;
}

// ---------- stage 3: GEMM1  [8192x1024]x[1024x256] + b1, ELU -> H1 bf16 ----------
// wave per (m-tile, 64-wide n-group); 8 waves/block. waves = 512*4 = 2048.
__global__ void gemm1_kernel(const __bf16* __restrict__ A, const __bf16* __restrict__ Bt,
                             const float* __restrict__ bias, __bf16* __restrict__ out) {
  const int K = C_IN, N = C1_;
  const int wid  = blockIdx.x * 8 + (threadIdx.x >> 5);
  const int lane = threadIdx.x & 31;
  const int tm = wid >> 2;        // / (N/64 == 4)
  const int ng = wid & 3;
  v8f acc[4] = {};
  wmma_16x64(A + (size_t)tm * 16 * K, Bt + (size_t)ng * 64 * K, K, lane, acc);
  const int hf = lane >> 4, nl = lane & 15;
#pragma unroll
  for (int j = 0; j < 4; ++j) {
    const int n = ng * 64 + j * 16 + nl;
    const float bv = bias[n];
#pragma unroll
    for (int v = 0; v < 8; ++v) {
      const int m = tm * 16 + v + 8 * hf;
      out[(size_t)m * N + n] = (__bf16)eluf(acc[j][v] + bv);
    }
  }
}

// ---------- stage 4: GEMM2  [8192x256]x[256x128] + b2, ELU -> H3 (scattered) ----------
// store H3[r][c2*16 + p] so GEMM3 can read it as [512][2048] row-major.
__global__ void gemm2_kernel(const __bf16* __restrict__ A, const __bf16* __restrict__ Bt,
                             const float* __restrict__ bias, __bf16* __restrict__ H3) {
  const int K = C1_;
  const int wid  = blockIdx.x * 8 + (threadIdx.x >> 5);
  const int lane = threadIdx.x & 31;
  const int tm = wid >> 1;        // / (128/64 == 2)
  const int ng = wid & 1;
  v8f acc[4] = {};
  wmma_16x64(A + (size_t)tm * 16 * K, Bt + (size_t)ng * 64 * K, K, lane, acc);
  const int hf = lane >> 4, nl = lane & 15;
#pragma unroll
  for (int j = 0; j < 4; ++j) {
    const int n = ng * 64 + j * 16 + nl;
    const float bv = bias[n];
#pragma unroll
    for (int v = 0; v < 8; ++v) {
      const int m  = tm * 16 + v + 8 * hf;
      const int r  = m >> 4;
      const int pp = m & 15;
      H3[(size_t)r * K3 + n * 16 + pp] = (__bf16)eluf(acc[j][v] + bv);
    }
  }
}

// ---------- stage 5: GEMM3  [512x2048]x[2048x2048] + bl, ELU -> f32 out ----------
__global__ void gemm3_kernel(const __bf16* __restrict__ A, const __bf16* __restrict__ Bt,
                             const float* __restrict__ bias, float* __restrict__ out) {
  const int K = K3, N = OUTD;
  const int wid  = blockIdx.x * 8 + (threadIdx.x >> 5);
  const int lane = threadIdx.x & 31;
  const int tm = wid >> 5;        // / (2048/64 == 32)
  const int ng = wid & 31;
  v8f acc[4] = {};
  wmma_16x64(A + (size_t)tm * 16 * K, Bt + (size_t)ng * 64 * K, K, lane, acc);
  const int hf = lane >> 4, nl = lane & 15;
#pragma unroll
  for (int j = 0; j < 4; ++j) {
    const int n = ng * 64 + j * 16 + nl;
    const float bv = bias[n];
#pragma unroll
    for (int v = 0; v < 8; ++v) {
      const int m = tm * 16 + v + 8 * hf;
      out[(size_t)m * N + n] = eluf(acc[j][v] + bv);
    }
  }
}

// ---------- launcher ----------
extern "C" void kernel_launch(void* const* d_in, const int* in_sizes, int n_in,
                              void* d_out, int out_size, void* d_ws, size_t ws_size,
                              hipStream_t stream) {
  const float* x     = (const float*)d_in[0];
  const float* boxes = (const float*)d_in[1];
  const float* scale = (const float*)d_in[2];
  const float* w1    = (const float*)d_in[3];
  const float* b1    = (const float*)d_in[4];
  const float* w2    = (const float*)d_in[5];
  const float* b2    = (const float*)d_in[6];
  const float* wl    = (const float*)d_in[7];
  const float* bl    = (const float*)d_in[8];
  float* out = (float*)d_out;

  // workspace layout (bytes)
  char* ws = (char*)d_ws;
  const size_t SZ_XT  = (size_t)BATCH * FH * FW * C_IN * 2;  // 64 MB
  const size_t SZ_A1  = (size_t)M1 * C_IN * 2;               // 16 MB
  const size_t SZ_H1  = (size_t)M1 * C1_ * 2;                // 4 MB
  const size_t SZ_H3  = (size_t)NROI * K3 * 2;               // 2 MB
  const size_t SZ_W1  = (size_t)C1_ * C_IN * 2;
  const size_t SZ_W2  = (size_t)C2_ * C1_ * 2;
  __bf16* xt  = (__bf16*)(ws);
  __bf16* A1  = (__bf16*)(ws + SZ_XT);
  __bf16* H1  = (__bf16*)(ws + SZ_XT + SZ_A1);
  __bf16* H3  = (__bf16*)(ws + SZ_XT + SZ_A1 + SZ_H1);
  __bf16* w1b = (__bf16*)(ws + SZ_XT + SZ_A1 + SZ_H1 + SZ_H3);
  __bf16* w2b = (__bf16*)(ws + SZ_XT + SZ_A1 + SZ_H1 + SZ_H3 + SZ_W1);
  __bf16* wlb = (__bf16*)(ws + SZ_XT + SZ_A1 + SZ_H1 + SZ_H3 + SZ_W1 + SZ_W2);

  // weight conversion (f32 -> bf16, [N][K] row-major kept as-is)
  cvt_bf16_kernel<<<(C1_ * C_IN + 255) / 256, 256, 0, stream>>>(w1, w1b, C1_ * C_IN);
  cvt_bf16_kernel<<<(C2_ * C1_  + 255) / 256, 256, 0, stream>>>(w2, w2b, C2_ * C1_);
  cvt_bf16_kernel<<<(OUTD * K3  + 255) / 256, 256, 0, stream>>>(wl, wlb, OUTD * K3);

  // NCHW f32 -> NHWC bf16
  transpose_nhwc_kernel<<<dim3(FW / 32, C_IN / 32, BATCH * FH), dim3(32, 8), 0, stream>>>(x, xt);

  // ROI align -> A1[8192][1024]
  roi_align_kernel<<<M1, 256, 0, stream>>>(xt, boxes, scale, A1);

  // GEMM chain (waves: 2048 / 1024 / 1024; 8 waves per 256-thread block)
  gemm1_kernel<<<256, 256, 0, stream>>>(A1, w1b, b1, H1);
  gemm2_kernel<<<128, 256, 0, stream>>>(H1, w2b, b2, H3);
  gemm3_kernel<<<128, 256, 0, stream>>>(H3, wlb, bl, out);
}
